// InteractLayer_83674552861142
// MI455X (gfx1250) — matrix operand order repaired and checked by reading, not compiled, not run on hardware
//
#include <hip/hip_runtime.h>

typedef __attribute__((ext_vector_type(16))) _Float16 v16h;
typedef __attribute__((ext_vector_type(8)))  float    v8f;
typedef __attribute__((ext_vector_type(4)))  float    v4f;
typedef __attribute__((ext_vector_type(4)))  unsigned u32x4;
typedef unsigned long long u64;
typedef unsigned short u16;

#define WPB  2      // waves per block (64 KB LDS total)
#define NBLK 1024

union FragH {
    v16h  v;
    u64   q[4];
    u32x4 x[2];
};

union PkH { _Float16 h[2]; unsigned u; };

// One wave = one batch:
//   Q/K/V = X(32x64) @ W(64x64), R^T = W_r^T @ X^T   (16x16x32 f16 WMMA, f32 accum)
//   per head h: S = Q_h K_h^T, softmax rows, O^T = V_h^T P^T + R_h^T, relu
// O^T form makes each lane's 8 outputs contiguous -> global_store_b128.
__global__ __launch_bounds__(WPB * 32) void interact_attn_kernel(
    const float* __restrict__ X,
    const float* __restrict__ Wq,
    const float* __restrict__ Wk,
    const float* __restrict__ Wv,
    const float* __restrict__ Wr,
    float* __restrict__ Out,
    int nBatch)
{
    __shared__ __attribute__((aligned(16))) _Float16 WT[4][64][64];   // W^T f16, 32 KB
    __shared__ __attribute__((aligned(16))) _Float16 XS[WPB][2048];   // X f16, reused as P
    __shared__ __attribute__((aligned(16))) _Float16 QS[WPB][2048];
    __shared__ __attribute__((aligned(16))) _Float16 KS[WPB][2048];
    __shared__ __attribute__((aligned(16))) _Float16 VT[WPB][2048];   // V^T [64][32]

    const int tid  = threadIdx.x;
    const int lane = tid & 31;
    const int wave = tid >> 5;
    const int hl   = lane >> 4;   // half-wave select
    const int ln   = lane & 15;

    // ---- stage weights transposed + f16 (once per block) ----
    const float* Ws[4] = {Wq, Wk, Wv, Wr};
    #pragma unroll
    for (int m = 0; m < 4; ++m) {
        const float* W = Ws[m];
        for (int i = tid; i < 64 * 64; i += WPB * 32) {
            WT[m][i & 63][i >> 6] = (_Float16)W[i];   // W[d][h] -> WT[h][d]
        }
    }
    __syncthreads();

    _Float16* xs = XS[wave];
    _Float16* qs = QS[wave];
    _Float16* ks = KS[wave];
    _Float16* vt = VT[wave];

    const int wgid = blockIdx.x * WPB + wave;
    const int nwav = NBLK * WPB;

    for (int b = wgid; b < nBatch; b += nwav) {
        const float* xb = X + (size_t)b * 2048;

        // speculative prefetch of next batch's X (global_prefetch path)
        if (b + nwav < nBatch) {
            const float* xn = X + (size_t)(b + nwav) * 2048;
            __builtin_prefetch(xn + lane * 64, 0, 0);
        }

        // ---- stage X (32x64 f32) as f16 row-major ----
        #pragma unroll
        for (int i = 0; i < 16; ++i) {
            v4f xv = *(const v4f*)(xb + (i * 32 + lane) * 4);
            union { _Float16 h[4]; u64 q; } t;
            t.h[0] = (_Float16)xv[0]; t.h[1] = (_Float16)xv[1];
            t.h[2] = (_Float16)xv[2]; t.h[3] = (_Float16)xv[3];
            *(u64*)(xs + (i * 32 + lane) * 4) = t.q;
        }

        // ---- A fragments of X (rows) ----
        FragH Ax[2][2];
        #pragma unroll
        for (int mi = 0; mi < 2; ++mi)
            #pragma unroll
            for (int kk = 0; kk < 2; ++kk) {
                const _Float16* p = xs + (mi * 16 + ln) * 64 + kk * 32 + hl * 8;
                Ax[mi][kk].q[0] = *(const u64*)(p);
                Ax[mi][kk].q[1] = *(const u64*)(p + 4);
                Ax[mi][kk].q[2] = *(const u64*)(p + 16);
                Ax[mi][kk].q[3] = *(const u64*)(p + 20);
            }

        // ---- R^T = W_r^T @ X^T : Racc[col tile 0..3][query tile 0..1] ----
        v8f Racc[4][2];
        {
            FragH Bx[2][2];   // B-fragments of X^T (lane = query, contiguous d)
            #pragma unroll
            for (int nq = 0; nq < 2; ++nq)
                #pragma unroll
                for (int kk = 0; kk < 2; ++kk) {
                    const _Float16* p = xs + (nq * 16 + ln) * 64 + kk * 32 + hl * 8;
                    Bx[nq][kk].x[0] = *(const u32x4*)(p);
                    Bx[nq][kk].x[1] = *(const u32x4*)(p + 16);
                }
            #pragma unroll
            for (int mc = 0; mc < 4; ++mc) {
                FragH A0, A1;
                const _Float16* p = &WT[3][0][0] + (mc * 16 + ln) * 64 + hl * 8;
                A0.q[0] = *(const u64*)(p);      A0.q[1] = *(const u64*)(p + 4);
                A0.q[2] = *(const u64*)(p + 16); A0.q[3] = *(const u64*)(p + 20);
                A1.q[0] = *(const u64*)(p + 32); A1.q[1] = *(const u64*)(p + 36);
                A1.q[2] = *(const u64*)(p + 48); A1.q[3] = *(const u64*)(p + 52);
                #pragma unroll
                for (int nq = 0; nq < 2; ++nq) {
                    v8f acc = {0.f, 0.f, 0.f, 0.f, 0.f, 0.f, 0.f, 0.f};
                    acc = __builtin_amdgcn_wmma_f32_16x16x32_f16(
                        false, A0.v, false, Bx[nq][0].v, (short)0, acc, false, false);
                    acc = __builtin_amdgcn_wmma_f32_16x16x32_f16(
                        false, A1.v, false, Bx[nq][1].v, (short)0, acc, false, false);
                    Racc[mc][nq] = acc;
                }
            }
        }

        // ---- projections Q, K, V ----
        #pragma unroll
        for (int m = 0; m < 3; ++m) {
            #pragma unroll
            for (int nj = 0; nj < 4; ++nj) {
                const _Float16* wrow = &WT[m][0][0] + (nj * 16 + ln) * 64 + hl * 8;
                FragH b0, b1;
                b0.x[0] = *(const u32x4*)(wrow);
                b0.x[1] = *(const u32x4*)(wrow + 16);
                b1.x[0] = *(const u32x4*)(wrow + 32);
                b1.x[1] = *(const u32x4*)(wrow + 48);
                #pragma unroll
                for (int mi = 0; mi < 2; ++mi) {
                    v8f acc = {0.f, 0.f, 0.f, 0.f, 0.f, 0.f, 0.f, 0.f};
                    acc = __builtin_amdgcn_wmma_f32_16x16x32_f16(
                        false, Ax[mi][0].v, false, b0.v, (short)0, acc, false, false);
                    acc = __builtin_amdgcn_wmma_f32_16x16x32_f16(
                        false, Ax[mi][1].v, false, b1.v, (short)0, acc, false, false);
                    if (m < 2) {
                        // Q/K row-major; packed cvt + b16 lo / b16 d16_hi stores
                        _Float16* dst = (m == 0 ? qs : ks) + (mi * 16 + hl * 8) * 64 + nj * 16 + ln;
                        #pragma unroll
                        for (int r = 0; r < 8; r += 2) {
                            PkH t2;
                            t2.h[0] = (_Float16)acc[r];
                            t2.h[1] = (_Float16)acc[r + 1];
                            *(u16*)(dst + r * 64)       = (u16)t2.u;
                            *(u16*)(dst + (r + 1) * 64) = (u16)(t2.u >> 16);
                        }
                    } else {
                        // V transposed, packed b32 stores
                        _Float16* dst = vt + (nj * 16 + ln) * 32 + mi * 16 + hl * 8;
                        #pragma unroll
                        for (int r = 0; r < 8; r += 2) {
                            PkH t2;
                            t2.h[0] = (_Float16)acc[r];
                            t2.h[1] = (_Float16)acc[r + 1];
                            *(unsigned*)(dst + r) = t2.u;
                        }
                    }
                }
            }
        }

        // ---- attention per head ----
        #pragma unroll
        for (int hh = 0; hh < 2; ++hh) {
            FragH Aq[2], Bk[2];
            #pragma unroll
            for (int mi = 0; mi < 2; ++mi) {
                const _Float16* p = qs + (mi * 16 + ln) * 64 + hh * 32 + hl * 8;
                Aq[mi].q[0] = *(const u64*)(p);
                Aq[mi].q[1] = *(const u64*)(p + 4);
                Aq[mi].q[2] = *(const u64*)(p + 16);
                Aq[mi].q[3] = *(const u64*)(p + 20);
            }
            #pragma unroll
            for (int nj = 0; nj < 2; ++nj) {
                const _Float16* p = ks + (nj * 16 + ln) * 64 + hh * 32 + hl * 8;
                Bk[nj].x[0] = *(const u32x4*)(p);
                Bk[nj].x[1] = *(const u32x4*)(p + 16);
            }

            v8f S[2][2];
            #pragma unroll
            for (int mi = 0; mi < 2; ++mi)
                #pragma unroll
                for (int nj = 0; nj < 2; ++nj) {
                    v8f z = {0.f, 0.f, 0.f, 0.f, 0.f, 0.f, 0.f, 0.f};
                    S[mi][nj] = __builtin_amdgcn_wmma_f32_16x16x32_f16(
                        false, Aq[mi].v, false, Bk[nj].v, (short)0, z, false, false);
                }

            // ---- softmax over rows (16-lane half-wave reductions) ----
            _Float16* pb = xs + hh * 1024;   // reuse X buffer for P (32x32 f16)
            #pragma unroll
            for (int mi = 0; mi < 2; ++mi) {
                #pragma unroll
                for (int r = 0; r < 8; ++r) {
                    float a0 = S[mi][0][r], a1 = S[mi][1][r];
                    float mx = fmaxf(a0, a1);
                    mx = fmaxf(mx, __shfl_xor(mx, 1));
                    mx = fmaxf(mx, __shfl_xor(mx, 2));
                    mx = fmaxf(mx, __shfl_xor(mx, 4));
                    mx = fmaxf(mx, __shfl_xor(mx, 8));
                    float e0 = __expf(a0 - mx);
                    float e1 = __expf(a1 - mx);
                    float s = e0 + e1;
                    s += __shfl_xor(s, 1);
                    s += __shfl_xor(s, 2);
                    s += __shfl_xor(s, 4);
                    s += __shfl_xor(s, 8);
                    float inv = __builtin_amdgcn_rcpf(s);
                    PkH tp;
                    tp.h[0] = (_Float16)(e0 * inv);
                    tp.h[1] = (_Float16)(e1 * inv);
                    _Float16* dp = pb + (mi * 16 + hl * 8 + r) * 32 + ln;
                    *(u16*)(dp)      = (u16)tp.u;          // col tile 0
                    *(u16*)(dp + 16) = (u16)(tp.u >> 16);  // col tile 1 (d16_hi)
                }
            }

            // ---- O^T = V_h^T @ P^T + R_h^T ; relu ; contiguous b128 stores ----
            #pragma unroll
            for (int mi = 0; mi < 2; ++mi) {          // column tile within head
                FragH Av;                              // A-frag of V_h^T from VT
                const _Float16* p = vt + (hh * 32 + mi * 16 + ln) * 32 + hl * 8;
                Av.q[0] = *(const u64*)(p);
                Av.q[1] = *(const u64*)(p + 4);
                Av.q[2] = *(const u64*)(p + 16);
                Av.q[3] = *(const u64*)(p + 20);
                #pragma unroll
                for (int nq = 0; nq < 2; ++nq) {      // query tile
                    FragH Bp;                          // B-frag of P^T (lane = query)
                    const _Float16* pp = pb + (nq * 16 + ln) * 32 + hl * 8;
                    Bp.x[0] = *(const u32x4*)(pp);
                    Bp.x[1] = *(const u32x4*)(pp + 16);
                    v8f o = __builtin_amdgcn_wmma_f32_16x16x32_f16(
                        false, Av.v, false, Bp.v, (short)0,
                        Racc[hh * 2 + mi][nq], false, false);
                    // lane: f = nq*16+ln, cols = hh*32+mi*16+hl*8 + (0..7) contiguous
                    float* op = Out + ((size_t)b * 32 + nq * 16 + ln) * 64
                                    + hh * 32 + mi * 16 + hl * 8;
                    v4f o0, o1;
                    o0[0] = fmaxf(o[0], 0.f); o0[1] = fmaxf(o[1], 0.f);
                    o0[2] = fmaxf(o[2], 0.f); o0[3] = fmaxf(o[3], 0.f);
                    o1[0] = fmaxf(o[4], 0.f); o1[1] = fmaxf(o[5], 0.f);
                    o1[2] = fmaxf(o[6], 0.f); o1[3] = fmaxf(o[7], 0.f);
                    *(v4f*)(op)     = o0;
                    *(v4f*)(op + 4) = o1;
                }
            }
        }
    }
}

extern "C" void kernel_launch(void* const* d_in, const int* in_sizes, int n_in,
                              void* d_out, int out_size, void* d_ws, size_t ws_size,
                              hipStream_t stream) {
    const float* X  = (const float*)d_in[0];
    const float* Wq = (const float*)d_in[1];
    const float* Wk = (const float*)d_in[2];
    const float* Wv = (const float*)d_in[3];
    const float* Wr = (const float*)d_in[4];
    float* Out = (float*)d_out;
    int nBatch = in_sizes[0] / (32 * 64);
    interact_attn_kernel<<<NBLK, WPB * 32, 0, stream>>>(X, Wq, Wk, Wv, Wr, Out, nBatch);
}
